// Sparse4DHead2nd_88691074662744
// MI455X (gfx1250) — compile-verified, matrix operand order checked by reading, not run on hardware
//
#include <hip/hip_runtime.h>

// ---------------- problem constants (match reference) ----------------
#define NA     900
#define MPAD   912       // NA rounded to 16
#define NTEMP  600
#define EDIM   256
#define NCAM   6
#define NLVL   4
#define NPT    13
#define NGRP   8
#define NLEARNK 6
#define NDEC   6
#define ADIM   11
#define NCLS   10
#define WTOT   2496      // NCAM*NLVL*NPT*NGRP
#define WTOTP  2560      // WTOT padded to 128
#define TSP    14872
#define CPG    32        // EDIM / NGRP
#define NTL    128       // GEMM N tile (4 waves x 2 subtiles x 16)

__device__ __constant__ float FIXC[7][3] = {
    {0.f,0.f,0.f},{0.5f,0.f,0.f},{-0.5f,0.f,0.f},
    {0.f,0.5f,0.f},{0.f,-0.5f,0.f},{0.f,0.f,0.5f},{0.f,0.f,-0.5f}};
__device__ __constant__ int SHC[4][2] = {{64,176},{32,88},{16,44},{8,22}};
__device__ __constant__ int LSIC[4]   = {0,11264,14080,14784};

typedef __attribute__((ext_vector_type(16))) _Float16 v16h;
typedef __attribute__((ext_vector_type(8)))  float    v8f;

union FragU { uint4 u[2]; v16h h; };
union PackU { _Float16 h[4]; uint2 u; };

// ---- weight pre-pack: f32 [K,N] -> f16 TRANSPOSED [Np,Kp], zero padded ----
__global__ void pack_bT(const float* __restrict__ B, _Float16* __restrict__ Bp,
                        int K, int N, int Kp, int total)
{
    int i = blockIdx.x * blockDim.x + threadIdx.x;
    if (i >= total) return;
    int n = i / Kp, k = i % Kp;
    Bp[i] = (k < K && n < N) ? (_Float16)B[(size_t)k * N + n] : (_Float16)0.f;
}

// anchor [NA,11] f32 -> f16 row-major [MPAD,32] zero padded in K
__global__ void pack_a16(const float* __restrict__ A, _Float16* __restrict__ Ap)
{
    int i = blockIdx.x * blockDim.x + threadIdx.x;
    if (i >= NA * 32) return;
    int r = i / 32, k = i % 32;
    Ap[i] = (k < ADIM) ? (_Float16)A[(size_t)r * ADIM + k] : (_Float16)0.f;
}

// ---------------- LDS double-buffered WMMA GEMM ----------------
// C[M,N] = op(A)[.,K] @ B + bias;  B pre-packed f16 transposed [Np,Kp].
// MODE 0: A = A1 f32 (stride K)      MODE 1: A = A1 + A2 (f32, stride K)
// MODE 2: A = concat(A1,A2) f32 (stride 256 each, K = 512)
// MODE 3: A = A16 f16 row-major (stride K = 32) -- fragments read from global
// 4 waves/block, each wave owns TWO 16x16 N-subtiles (cols wv*16 and +64),
// sharing one LDS A tile; A staging is double-buffered (one barrier / k-step).
template<int MODE, int RELU>
__global__ __launch_bounds__(128) void gemm_wmma_tiled(
    const float* __restrict__ A1, const float* __restrict__ A2,
    const _Float16* __restrict__ A16,
    const _Float16* __restrict__ BpT, const float* __restrict__ bias,
    float* __restrict__ C, int M, int K, int N)
{
    __shared__ _Float16 lsA[2][16 * 32];

    const int t    = threadIdx.x;
    const int mt   = blockIdx.x, nt = blockIdx.y;
    const int wv   = t >> 5;
    const int lane = t & 31;
    const int half = lane >> 4;
    const int l15  = lane & 15;
    const int col0 = nt * NTL + wv * 16 + l15;          // padded-N columns
    const _Float16* brow0 = BpT + (size_t)col0 * K;
    const _Float16* brow1 = brow0 + (size_t)64 * K;     // second subtile (+64)

    v8f acc0 = {}, acc1 = {};

    if (MODE == 3) {
        for (int kb = 0; kb < K; kb += 32) {
            FragU fa, fb0, fb1;
            const _Float16* arow = A16 + (size_t)(mt * 16 + l15) * K + kb;
            fa.u[0]  = *(const uint4*)(arow + 8 * half);
            fa.u[1]  = *(const uint4*)(arow + 16 + 8 * half);
            fb0.u[0] = *(const uint4*)(brow0 + kb + 8 * half);
            fb0.u[1] = *(const uint4*)(brow0 + kb + 16 + 8 * half);
            fb1.u[0] = *(const uint4*)(brow1 + kb + 8 * half);
            fb1.u[1] = *(const uint4*)(brow1 + kb + 16 + 8 * half);
            acc0 = __builtin_amdgcn_wmma_f32_16x16x32_f16(
                       false, fa.h, false, fb0.h, (short)0, acc0, false, false);
            acc1 = __builtin_amdgcn_wmma_f32_16x16x32_f16(
                       false, fa.h, false, fb1.h, (short)0, acc1, false, false);
        }
    } else {
        const int row = t >> 3, seg = t & 7;            // 16 rows x 8 segs of 4
        // A-tile loader for k-block starting at kb
        auto loadA = [&](int kb) -> float4 {
            const int k = kb + seg * 4;
            float4 v;
            if (MODE == 2) {
                const float* src = (k < 256)
                    ? (A1 + (size_t)(mt * 16 + row) * 256 + k)
                    : (A2 + (size_t)(mt * 16 + row) * 256 + (k - 256));
                v = *(const float4*)src;
            } else {
                const float* src = A1 + (size_t)(mt * 16 + row) * K + k;
                v = *(const float4*)src;
                if (MODE == 1) {
                    float4 v2 = *(const float4*)(A2 + (size_t)(mt * 16 + row) * K + k);
                    v.x += v2.x; v.y += v2.y; v.z += v2.z; v.w += v2.w;
                }
            }
            return v;
        };

        float4 v = loadA(0);
        int buf = 0;
        for (int kb = 0; kb < K; kb += 32) {
            PackU pk;
            pk.h[0] = (_Float16)v.x; pk.h[1] = (_Float16)v.y;
            pk.h[2] = (_Float16)v.z; pk.h[3] = (_Float16)v.w;
            *(uint2*)&lsA[buf][row * 32 + seg * 4] = pk.u;   // ds_store_b64
            __syncthreads();
            if (kb + 32 < K) v = loadA(kb + 32);             // overlaps compute
            FragU fa, fb0, fb1;
            fa.u[0]  = *(const uint4*)&lsA[buf][l15 * 32 + 8 * half];       // ds b128
            fa.u[1]  = *(const uint4*)&lsA[buf][l15 * 32 + 16 + 8 * half];  // ds b128
            fb0.u[0] = *(const uint4*)(brow0 + kb + 8 * half);       // global b128
            fb0.u[1] = *(const uint4*)(brow0 + kb + 16 + 8 * half);
            fb1.u[0] = *(const uint4*)(brow1 + kb + 8 * half);
            fb1.u[1] = *(const uint4*)(brow1 + kb + 16 + 8 * half);
            acc0 = __builtin_amdgcn_wmma_f32_16x16x32_f16(
                       false, fa.h, false, fb0.h, (short)0, acc0, false, false);
            acc1 = __builtin_amdgcn_wmma_f32_16x16x32_f16(
                       false, fa.h, false, fb1.h, (short)0, acc1, false, false);
            buf ^= 1;
        }
    }

    const float bv0 = (bias && col0 < N)      ? bias[col0]      : 0.f;
    const float bv1 = (bias && col0 + 64 < N) ? bias[col0 + 64] : 0.f;
#pragma unroll
    for (int r = 0; r < 8; ++r) {
        const int m = mt * 16 + r + 8 * half;   // C/D: M = vgpr + 8*half
        if (m < M) {
            if (col0 < N) {
                float v = acc0[r] + bv0;
                if (RELU) v = fmaxf(v, 0.f);
                C[(size_t)m * N + col0] = v;
            }
            if (col0 + 64 < N) {
                float v = acc1[r] + bv1;
                if (RELU) v = fmaxf(v, 0.f);
                C[(size_t)m * N + col0 + 64] = v;
            }
        }
    }
}

// ---------------- keypoints + projection ----------------
__global__ void kps_project(const float* __restrict__ anchor,
                            const float* __restrict__ learn,
                            const float* __restrict__ l2i,
                            const float* __restrict__ image_wh,
                            float* __restrict__ uv)
{
    int idx = blockIdx.x * blockDim.x + threadIdx.x;
    if (idx >= NA * NPT * NCAM) return;
    int c  = idx % NCAM;
    int ap = idx / NCAM;
    int p  = ap % NPT;
    int a  = ap / NPT;

    const float* an = anchor + (size_t)a * ADIM;
    float cx = an[0], cy = an[1], cz = an[2];
    float sx = expf(an[3]), sy = expf(an[4]), sz = expf(an[5]);
    float kx, ky, kz;
    if (p < 7) {
        kx = cx + FIXC[p][0] * sx; ky = cy + FIXC[p][1] * sy; kz = cz + FIXC[p][2] * sz;
    } else {
        const float* l = learn + (size_t)a * (NLEARNK * 3) + (p - 7) * 3;
        kx = cx + l[0] * sx; ky = cy + l[1] * sy; kz = cz + l[2] * sz;
    }
    const float* M = l2i + (size_t)c * 16;
    float p0 = M[0] * kx + M[1] * ky + M[2]  * kz + M[3];
    float p1 = M[4] * kx + M[5] * ky + M[6]  * kz + M[7];
    float p2 = M[8] * kx + M[9] * ky + M[10] * kz + M[11];
    p2 = fmaxf(p2, 1e-5f);
    uv[(size_t)idx * 2 + 0] = (p0 / p2) / image_wh[c * 2 + 0];
    uv[(size_t)idx * 2 + 1] = (p1 / p2) / image_wh[c * 2 + 1];
}

// ---------------- grouped softmax over NC*NL*NP axis ----------------
__global__ void softmax_groups(float* __restrict__ w)
{
    int t = blockIdx.x * blockDim.x + threadIdx.x;
    if (t >= NA * NGRP) return;
    int a = t / NGRP, g = t % NGRP;
    float* base = w + (size_t)a * WTOT + g;
    const int J = WTOT / NGRP;    // 312
    float mx = -1e30f;
    for (int j = 0; j < J; ++j) mx = fmaxf(mx, base[j * NGRP]);
    float s = 0.f;
    for (int j = 0; j < J; ++j) s += expf(base[j * NGRP] - mx);
    float inv = 1.f / s;
    for (int j = 0; j < J; ++j) base[j * NGRP] = expf(base[j * NGRP] - mx) * inv;
}

// ---------------- deformable aggregation (bilinear gather) ----------------
__global__ __launch_bounds__(256) void daf_kernel(
    const float* __restrict__ feature, const float* __restrict__ uv,
    const float* __restrict__ w, float* __restrict__ out,
    float* __restrict__ out2)
{
    const int a  = blockIdx.x;
    const int ch = threadIdx.x;
    const int g  = ch / CPG;
    float acc = 0.f;

    for (int l = 0; l < NLVL; ++l) {
        const int H = SHC[l][0], W = SHC[l][1], st = LSIC[l];
        for (int p = 0; p < NPT; ++p) {
            for (int c = 0; c < NCAM; ++c) {
                const int ui = (((a * NPT) + p) * NCAM + c) * 2;
                const float x = uv[ui]     * (float)W - 0.5f;
                const float y = uv[ui + 1] * (float)H - 0.5f;
                const float x0 = floorf(x), y0 = floorf(y);
                const float dx = x - x0,   dy = y - y0;
                const float wt = w[(size_t)a * WTOT + (((c * NLVL + l) * NPT + p) * NGRP) + g];
                const float* fb = feature + (size_t)c * TSP * EDIM + ch;

                const float xs[4] = {x0, x0 + 1.f, x0,       x0 + 1.f};
                const float ys[4] = {y0, y0,       y0 + 1.f, y0 + 1.f};
                const float bw[4] = {(1.f - dx) * (1.f - dy), dx * (1.f - dy),
                                     (1.f - dx) * dy,          dx * dy};
                {
                    int xi = (int)fminf(fmaxf(x0 + 1.f, 0.f), (float)(W - 1));
                    int yi = (int)fminf(fmaxf(y0 + 1.f, 0.f), (float)(H - 1));
                    __builtin_prefetch((const void*)(fb + (size_t)(st + yi * W + xi) * EDIM), 0, 1);
                }
#pragma unroll
                for (int tp = 0; tp < 4; ++tp) {
                    const float cxx = xs[tp], cyy = ys[tp];
                    const bool valid = (cxx >= 0.f) && (cxx < (float)W) &&
                                       (cyy >= 0.f) && (cyy < (float)H);
                    const int xi = (int)fminf(fmaxf(cxx, 0.f), (float)(W - 1));
                    const int yi = (int)fminf(fmaxf(cyy, 0.f), (float)(H - 1));
                    const float fv = fb[(size_t)(st + yi * W + xi) * EDIM];
                    acc += fv * bw[tp] * (valid ? 1.f : 0.f) * wt;
                }
            }
        }
    }
    out[(size_t)a * EDIM + ch]  = acc;   // padded ws buffer (GEMM A input)
    out2[(size_t)a * EDIM + ch] = acc;   // tmp_outs[i] slot in d_out
}

// ---------------- small helpers ----------------
__global__ void copy_f(const float* __restrict__ s, float* __restrict__ d, int n)
{
    int i = blockIdx.x * blockDim.x + threadIdx.x;
    if (i < n) d[i] = s[i];
}

__global__ void anchor_update(const float* __restrict__ acur, const float* __restrict__ reg,
                              const float* __restrict__ tint, float* __restrict__ anxt)
{
    int i = blockIdx.x * blockDim.x + threadIdx.x;
    if (i >= NA * ADIM) return;
    int j = i % ADIM;
    float r = reg[i];
    if (j >= 8) r /= tint[0];
    anxt[i] = acur[i] + r;
}

__global__ void score_max(const float* __restrict__ cls, float* __restrict__ scores)
{
    int a = blockIdx.x * blockDim.x + threadIdx.x;
    if (a >= NA) return;
    float m = -1e30f;
    for (int k = 0; k < NCLS; ++k) m = fmaxf(m, cls[(size_t)a * NCLS + k]);
    scores[a] = m;
}

__global__ __launch_bounds__(256) void topk300(const float* __restrict__ scores,
                                               int* __restrict__ idxs)
{
    __shared__ float s[NA];
    __shared__ float bval[256];
    __shared__ int   bidx[256];
    const int t = threadIdx.x;
    for (int i = t; i < NA; i += 256) s[i] = scores[i];
    __syncthreads();
    for (int k = 0; k < NA - NTEMP; ++k) {
        float bv = -1e30f; int bi = NA;
        for (int i = t; i < NA; i += 256) {
            float v = s[i];
            if (v > bv || (v == bv && i < bi)) { bv = v; bi = i; }
        }
        bval[t] = bv; bidx[t] = bi;
        __syncthreads();
        for (int off = 128; off > 0; off >>= 1) {
            if (t < off) {
                float ov = bval[t + off]; int oi = bidx[t + off];
                if (ov > bval[t] || (ov == bval[t] && oi < bidx[t])) { bval[t] = ov; bidx[t] = oi; }
            }
            __syncthreads();
        }
        if (t == 0) { idxs[k] = bidx[0]; s[bidx[0]] = -1e30f; }
        __syncthreads();
    }
}

__global__ void select_feat(const float* __restrict__ fnew, const float* __restrict__ tempf,
                            const int* __restrict__ idxs, const unsigned char* __restrict__ mask,
                            float* __restrict__ outf)
{
    int i = blockIdx.x * blockDim.x + threadIdx.x;
    if (i >= NA * EDIM) return;
    int a = i / EDIM, ch = i % EDIM;
    float v;
    if (mask[0]) v = (a < NTEMP) ? tempf[(size_t)a * EDIM + ch]
                                 : fnew[(size_t)idxs[a - NTEMP] * EDIM + ch];
    else v = fnew[i];
    outf[i] = v;
}

__global__ void select_anchor(const float* __restrict__ anew, const float* __restrict__ tempa,
                              const int* __restrict__ idxs, const unsigned char* __restrict__ mask,
                              float* __restrict__ outa)
{
    int i = blockIdx.x * blockDim.x + threadIdx.x;
    if (i >= NA * ADIM) return;
    int a = i / ADIM, j = i % ADIM;
    float v;
    if (mask[0]) v = (a < NTEMP) ? tempa[(size_t)a * ADIM + j]
                                 : anew[(size_t)idxs[a - NTEMP] * ADIM + j];
    else v = anew[i];
    outa[i] = v;
}

__global__ void track_out(const int* __restrict__ tid, const unsigned char* __restrict__ mask,
                          float* __restrict__ out)
{
    int a = blockIdx.x * blockDim.x + threadIdx.x;
    if (a >= NA) return;
    out[a] = (float)(mask[0] ? tid[a] : -1);
}

// ---------------- host orchestration ----------------
extern "C" void kernel_launch(void* const* d_in, const int* in_sizes, int n_in,
                              void* d_out, int out_size, void* d_ws, size_t ws_size,
                              hipStream_t stream)
{
    (void)in_sizes; (void)n_in; (void)out_size; (void)ws_size;

    const float* feature  = (const float*)d_in[0];
    const float* inst0    = (const float*)d_in[3];
    const float* anchor0  = (const float*)d_in[4];
    const float* tint     = (const float*)d_in[5];
    const float* tempf    = (const float*)d_in[6];
    const float* tempa    = (const float*)d_in[7];
    const unsigned char* mask = (const unsigned char*)d_in[8];
    const int*   track_id = (const int*)d_in[9];
    const float* image_wh = (const float*)d_in[10];
    const float* l2i      = (const float*)d_in[11];
    const float* anchor_w = (const float*)d_in[12];
    const float* anchor_b = (const float*)d_in[13];
    const float* kps_w    = (const float*)d_in[14];
    const float* kps_b    = (const float*)d_in[15];
    const float* wfc_w    = (const float*)d_in[16];
    const float* wfc_b    = (const float*)d_in[17];
    const float* outp_w   = (const float*)d_in[18];
    const float* outp_b   = (const float*)d_in[19];
    const float* ffn_w    = (const float*)d_in[20];
    const float* ffn_b    = (const float*)d_in[21];
    const float* reg_w    = (const float*)d_in[22];
    const float* reg_b    = (const float*)d_in[23];
    const float* cls_w    = (const float*)d_in[24];
    const float* cls_b    = (const float*)d_in[25];
    const float* qt_w     = (const float*)d_in[26];
    const float* qt_b     = (const float*)d_in[27];

    float* out = (float*)d_out;
    float* wsf = (float*)d_ws;

    // ---- f32 workspace (A-side buffers padded to MPAD rows) ----
    size_t o = 0;
    float* featA   = wsf + o; o += (size_t)MPAD * EDIM;
    float* featB   = wsf + o; o += (size_t)MPAD * EDIM;
    float* embed   = wsf + o; o += (size_t)MPAD * EDIM;
    float* out_tmp = wsf + o; o += (size_t)MPAD * EDIM;
    float* dafbuf  = wsf + o; o += (size_t)MPAD * EDIM;
    float* anchorA = wsf + o; o += (size_t)NA * ADIM;
    float* anchorB = wsf + o; o += (size_t)NA * ADIM;
    float* learn   = wsf + o; o += (size_t)NA * NLEARNK * 3;
    float* uvb     = wsf + o; o += (size_t)NA * NPT * NCAM * 2;
    float* wbuf    = wsf + o; o += (size_t)NA * WTOT;
    float* regb    = wsf + o; o += (size_t)NA * ADIM;
    float* scores  = wsf + o; o += NA;
    int*   idxs    = (int*)(wsf + o); o += (NA - NTEMP);

    // ---- f16 packed-weight workspace (transposed [Np,Kp], Np mult of 128) ----
    _Float16* h = (_Float16*)(wsf + o);
    size_t ho = 0;
    _Float16* a16buf = h + ho; ho += (size_t)MPAD * 32;
    _Float16* anch_p = h + ho; ho += 256 * 32;
    _Float16 *kps_p[NDEC], *wfc_p[NDEC], *outp_p[NDEC], *ffn_p[NDEC],
             *reg_p[NDEC], *cls_p[NDEC], *qt_p[NDEC];
    for (int i = 0; i < NDEC; ++i) {
        kps_p[i]  = h + ho; ho += 128 * 256;
        wfc_p[i]  = h + ho; ho += (size_t)WTOTP * 256;
        outp_p[i] = h + ho; ho += 256 * 256;
        ffn_p[i]  = h + ho; ho += 256 * 512;
        reg_p[i]  = h + ho; ho += 128 * 256;
        cls_p[i]  = h + ho; ho += 128 * 256;
        qt_p[i]   = h + ho; ho += 128 * 256;
    }

    // ---- output layout (reference return order, flattened) ----
    float* o_feat   = out;
    float* o_anchor = out + (size_t)NA * EDIM;
    float* o_cls    = o_anchor + (size_t)NA * ADIM;
    float* o_qt     = o_cls + (size_t)NA * NCLS;
    float* o_trk    = o_qt + NA;
    float* o_tmp    = o_trk + NA;

    auto packB = [&](const float* B, _Float16* Bp, int K, int N, int Kp, int Np) {
        int total = Np * Kp;
        pack_bT<<<(total + 255) / 256, 256, 0, stream>>>(B, Bp, K, N, Kp, total);
    };
    auto gemm = [&](int mode, int relu, const float* A1, const float* A2,
                    const _Float16* A16, const _Float16* Bp, const float* bias,
                    float* C, int M, int K, int Np, int N) {
        dim3 grid((M + 15) / 16, (Np + NTL - 1) / NTL);
        dim3 blk(128);
        (void)relu;
        if (mode == 0)      gemm_wmma_tiled<0,0><<<grid, blk, 0, stream>>>(A1, A2, A16, Bp, bias, C, M, K, N);
        else if (mode == 1) gemm_wmma_tiled<1,0><<<grid, blk, 0, stream>>>(A1, A2, A16, Bp, bias, C, M, K, N);
        else if (mode == 2) gemm_wmma_tiled<2,1><<<grid, blk, 0, stream>>>(A1, A2, A16, Bp, bias, C, M, K, N);
        else                gemm_wmma_tiled<3,0><<<grid, blk, 0, stream>>>(A1, A2, A16, Bp, bias, C, M, K, N);
    };

    // ---- pre-pack all weights to padded transposed f16 ----
    packB(anchor_w, anch_p, ADIM, 256, 32, 256);
    for (int i = 0; i < NDEC; ++i) {
        packB(kps_w  + (size_t)i * EDIM * 18,       kps_p[i],  256, 18,   256, 128);
        packB(wfc_w  + (size_t)i * EDIM * WTOT,     wfc_p[i],  256, WTOT, 256, WTOTP);
        packB(outp_w + (size_t)i * EDIM * EDIM,     outp_p[i], 256, 256,  256, 256);
        packB(ffn_w  + (size_t)i * 2 * EDIM * EDIM, ffn_p[i],  512, 256,  512, 256);
        packB(reg_w  + (size_t)i * EDIM * ADIM,     reg_p[i],  256, ADIM, 256, 128);
        packB(cls_w  + (size_t)i * EDIM * NCLS,     cls_p[i],  256, NCLS, 256, 128);
        packB(qt_w   + (size_t)i * EDIM,            qt_p[i],   256, 1,    256, 128);
    }

    copy_f<<<(NA * EDIM + 255) / 256, 256, 0, stream>>>(inst0, featA, NA * EDIM);
    copy_f<<<(NA * ADIM + 255) / 256, 256, 0, stream>>>(anchor0, anchorA, NA * ADIM);

    float* fcur = featA; float* fnxt = featB;
    float* acur = anchorA; float* anxt = anchorB;

    for (int i = 0; i < NDEC; ++i) {
        // anchor_embed = anchor @ anchor_w + anchor_b  (K padded 11 -> 32)
        pack_a16<<<(NA * 32 + 255) / 256, 256, 0, stream>>>(acur, a16buf);
        gemm(3, 0, nullptr, nullptr, a16buf, anch_p, anchor_b, embed, NA, 32, 256, 256);
        // learned keypoint offsets
        gemm(0, 0, fcur, nullptr, nullptr, kps_p[i], kps_b + (size_t)i * 18, learn,
             NA, 256, 128, 18);
        kps_project<<<(NA * NPT * NCAM + 255) / 256, 256, 0, stream>>>(acur, learn, l2i, image_wh, uvb);
        // attention weights: (feat + embed) @ wfc
        gemm(1, 0, fcur, embed, nullptr, wfc_p[i], wfc_b + (size_t)i * WTOT, wbuf,
             NA, 256, WTOTP, WTOT);
        softmax_groups<<<(NA * NGRP + 255) / 256, 256, 0, stream>>>(wbuf);
        // deformable aggregation -> padded ws buffer + tmp_outs[i]
        daf_kernel<<<NA, 256, 0, stream>>>(feature, uvb, wbuf, dafbuf,
                                           o_tmp + (size_t)i * NA * EDIM);
        // output proj, then ffn with concat(out, feat) + relu
        gemm(0, 0, dafbuf, nullptr, nullptr, outp_p[i], outp_b + (size_t)i * EDIM, out_tmp,
             NA, 256, 256, 256);
        gemm(2, 1, out_tmp, fcur, nullptr, ffn_p[i], ffn_b + (size_t)i * EDIM, fnxt,
             NA, 512, 256, 256);
        // heads
        gemm(0, 0, fnxt, nullptr, nullptr, reg_p[i], reg_b + (size_t)i * ADIM, regb,
             NA, 256, 128, ADIM);
        anchor_update<<<(NA * ADIM + 255) / 256, 256, 0, stream>>>(acur, regb, tint, anxt);
        gemm(0, 0, fnxt, nullptr, nullptr, cls_p[i], cls_b + (size_t)i * NCLS, o_cls,
             NA, 256, 128, NCLS);
        gemm(0, 0, fnxt, nullptr, nullptr, qt_p[i], qt_b + (size_t)i, o_qt,
             NA, 256, 128, 1);

        if (i == 0) {
            score_max<<<(NA + 255) / 256, 256, 0, stream>>>(o_cls, scores);
            topk300<<<1, 256, 0, stream>>>(scores, idxs);
            select_feat<<<(NA * EDIM + 255) / 256, 256, 0, stream>>>(fnxt, tempf, idxs, mask, fcur);
            select_anchor<<<(NA * ADIM + 255) / 256, 256, 0, stream>>>(anxt, tempa, idxs, mask, acur);
            track_out<<<(NA + 255) / 256, 256, 0, stream>>>(track_id, mask, o_trk);
            // selection wrote into fcur/acur: they remain "current" for i=1
        } else {
            float* t;
            t = fcur; fcur = fnxt; fnxt = t;
            t = acur; acur = anxt; anxt = t;
        }
    }

    copy_f<<<(NA * EDIM + 255) / 256, 256, 0, stream>>>(fcur, o_feat, NA * EDIM);
    copy_f<<<(NA * ADIM + 255) / 256, 256, 0, stream>>>(acur, o_anchor, NA * ADIM);
}
